// MultiHeadGeneticAttention_75307956568360
// MI455X (gfx1250) — matrix operand order repaired and checked by reading, not compiled, not run on hardware
//
#include <hip/hip_runtime.h>

// ---------------------------------------------------------------------------
// MultiHeadGeneticAttention for MI455X (gfx1250), wave32 + WMMA f16 (f32 acc)
// 32x64 register blocking per wave; ping-pong double-buffered fragment loads
// (no register rotation -> no v_mov shuffling / WMMA hazard NOPs).
// ---------------------------------------------------------------------------

#define D_   1024
#define T_   2048
#define NH_  16
#define HD_  64
#define HV_  4
#define NQK_ 2304   // D + D + HV*HD
#define WIN_ 512
#define EPS_ 1.1920929e-07f

typedef __attribute__((ext_vector_type(16))) _Float16 v16h;
typedef __attribute__((ext_vector_type(8)))  _Float16 v8h;
typedef __attribute__((ext_vector_type(8)))  float    v8f;

union AFrag { v16h v; v8h h[2]; };

__device__ __forceinline__ v8f wmma_f16(v16h a, v16h b, v8f c) {
  // D = A(16x32 f16) x B(32x16 f16) + C(16x16 f32)
  return __builtin_amdgcn_wmma_f32_16x16x32_f16(false, a, false, b, (short)0, c,
                                                false, false);
}

// A-fragment (16x32 MxK) from row-major f16; element (m,k) at p[m*ld + k].
__device__ __forceinline__ v16h load_a_rm(const _Float16* p, int ld) {
  int lane = threadIdx.x & 31;
  int m  = lane & 15;
  int kb = (lane >> 4) * 16;
  AFrag a;
  a.h[0] = *(const v8h*)(p + (size_t)m * ld + kb);
  a.h[1] = *(const v8h*)(p + (size_t)m * ld + kb + 8);
  return a.v;
}

// B-fragment (32x16 KxN) from "column-major" f16; element (k,n) at p[n*ld + k].
__device__ __forceinline__ v16h load_b_cm(const _Float16* p, int ld) {
  int lane = threadIdx.x & 31;
  int n  = lane & 15;
  int kb = (lane >> 4) * 16;
  return *(const v16h*)(p + (size_t)n * ld + kb);
}

// Butterfly sum across the 16-lane half-group (wave32).
__device__ __forceinline__ float hsum16(float r) {
#pragma unroll
  for (int off = 1; off < 16; off <<= 1) r += __shfl_xor(r, off, 32);
  return r;
}

// GEMM ping-pong helpers (pointers must be named xp0/xp1/wp in scope)
#define GEMM_LOAD(AA0, AA1, BB, OFFK)                                          \
  {                                                                            \
    AA0 = load_a_rm(xp0 + (OFFK), D_);                                         \
    AA1 = load_a_rm(xp1 + (OFFK), D_);                                         \
    _Pragma("unroll") for (int f = 0; f < 4; ++f)                              \
        BB[f] = load_b_cm(wp + (size_t)(f * 16) * D_ + (OFFK), D_);            \
  }
#define GEMM_STEP(AA0, AA1, BB)                                                \
  {                                                                            \
    _Pragma("unroll") for (int f = 0; f < 4; ++f)                              \
        acc[0][f] = wmma_f16(AA0, BB[f], acc[0][f]);                           \
    _Pragma("unroll") for (int f = 0; f < 4; ++f)                              \
        acc[1][f] = wmma_f16(AA1, BB[f], acc[1][f]);                           \
  }

// ---------------------------------------------------------------------------
// Kernel 0: f16 conversions: weights -> transposed [n][k]; x -> Xh row-major.
// Also zeroes per-head suminv.
// ---------------------------------------------------------------------------
__global__ void mhga_wcvt(const float* __restrict__ x,
                          const float* __restrict__ wq, const float* __restrict__ wk,
                          const float* __restrict__ wv, const float* __restrict__ wo,
                          _Float16* __restrict__ WhT, _Float16* __restrict__ WoT,
                          _Float16* __restrict__ Xh, float* __restrict__ suminv) {
  if (blockIdx.x == 0 && threadIdx.x < NH_) suminv[threadIdx.x] = 0.0f;
  const int total1 = NQK_ * D_;      // WhT
  const int total2 = D_ * D_;        // WoT
  const int total3 = T_ * D_;        // Xh
  for (int i = blockIdx.x * blockDim.x + threadIdx.x;
       i < total1 + total2 + total3; i += gridDim.x * blockDim.x) {
    if (i < total1) {
      int n = i / D_, k = i % D_;
      float v;
      if (n < D_)          v = wq[(size_t)k * D_ + n];
      else if (n < 2 * D_) v = wk[(size_t)k * D_ + (n - D_)];
      else                 v = wv[(size_t)k * (HV_ * HD_) + (n - 2 * D_)];
      WhT[i] = (_Float16)v;
    } else if (i < total1 + total2) {
      int j = i - total1;
      int n = j / D_, k = j % D_;
      WoT[j] = (_Float16)wo[(size_t)k * D_ + n];
    } else {
      int j = i - total1 - total2;
      Xh[j] = (_Float16)x[j];
    }
  }
}

// ---------------------------------------------------------------------------
// Kernel 1: fused QKV projection + bias + per-head RMSNorm, f16 outputs.
// Grid (NQK/64, T/128), 4 waves; each wave owns a 32x64 strip (2 M-tiles).
// Qh/Kh: [h][t][hd] f16.  Vt: [hv][hd][t] f16 (transposed for PV B-frags).
// ---------------------------------------------------------------------------
__global__ __launch_bounds__(128) void mhga_proj(
    const _Float16* __restrict__ Xh, const _Float16* __restrict__ WhT,
    const float* __restrict__ bq, const float* __restrict__ bk,
    const float* __restrict__ bv, const float* __restrict__ rqw,
    const float* __restrict__ rkw, _Float16* __restrict__ Qh,
    _Float16* __restrict__ Kh, _Float16* __restrict__ Vt) {
  int n0 = blockIdx.x * 64;
  int wave = threadIdx.x >> 5, lane = threadIdx.x & 31;
  int t0 = blockIdx.y * 128 + wave * 32;
  int nl = lane & 15, mb = (lane >> 4) * 8;

  const _Float16* xp0 = Xh + (size_t)t0 * D_;
  const _Float16* xp1 = Xh + (size_t)(t0 + 16) * D_;
  const _Float16* wp  = WhT + (size_t)n0 * D_;

  v8f acc[2][4];
#pragma unroll
  for (int mt = 0; mt < 2; ++mt)
#pragma unroll
    for (int f = 0; f < 4; ++f) acc[mt][f] = v8f{};

  // ping-pong: buffers P (even k-steps) and Q (odd k-steps), tail peeled
  v16h aP0, aP1, bP[4], aQ0, aQ1, bQ[4];
  GEMM_LOAD(aP0, aP1, bP, 0);
  int kk = 0;
  for (; kk < D_ - 64; kk += 64) {
    GEMM_LOAD(aQ0, aQ1, bQ, kk + 32);
    GEMM_STEP(aP0, aP1, bP);
    GEMM_LOAD(aP0, aP1, bP, kk + 64);
    GEMM_STEP(aQ0, aQ1, bQ);
  }
  GEMM_LOAD(aQ0, aQ1, bQ, kk + 32);
  GEMM_STEP(aP0, aP1, bP);
  GEMM_STEP(aQ0, aQ1, bQ);

  // bias (tile entirely inside one of Q/K/V: region bounds are 64-aligned)
  float bias[4];
#pragma unroll
  for (int f = 0; f < 4; ++f) {
    int n = n0 + f * 16 + nl;
    bias[f] = (n < D_) ? bq[n] : (n < 2 * D_) ? bk[n - D_] : bv[n - 2 * D_];
  }
#pragma unroll
  for (int mt = 0; mt < 2; ++mt)
#pragma unroll
    for (int f = 0; f < 4; ++f)
#pragma unroll
      for (int v = 0; v < 8; ++v) acc[mt][f][v] += bias[f];

  if (n0 < 2 * D_) {
    const float* rw = (n0 < D_) ? rqw : rkw;
    int h = (n0 < D_) ? (n0 >> 6) : ((n0 - D_) >> 6);
    _Float16* dst = ((n0 < D_) ? Qh : Kh) + (size_t)h * T_ * HD_;
#pragma unroll
    for (int mt = 0; mt < 2; ++mt) {
      float scale[8];
#pragma unroll
      for (int v = 0; v < 8; ++v) {
        float ss = 0.0f;
#pragma unroll
        for (int f = 0; f < 4; ++f) ss += acc[mt][f][v] * acc[mt][f][v];
        ss = hsum16(ss);
        scale[v] = rsqrtf(ss * (1.0f / HD_) + EPS_);
      }
#pragma unroll
      for (int f = 0; f < 4; ++f) {
        int hd = f * 16 + nl;
        float w = rw[hd];
#pragma unroll
        for (int v = 0; v < 8; ++v) {
          int t = t0 + mt * 16 + mb + v;
          dst[(size_t)t * HD_ + hd] = (_Float16)(acc[mt][f][v] * scale[v] * w);
        }
      }
    }
  } else {
    int hv = (n0 - 2 * D_) >> 6;
    _Float16* dst = Vt + (size_t)hv * HD_ * T_;
#pragma unroll
    for (int mt = 0; mt < 2; ++mt)
#pragma unroll
      for (int f = 0; f < 4; ++f) {
        int hd = f * 16 + nl;
#pragma unroll
        for (int v = 0; v < 8; ++v) {
          int t = t0 + mt * 16 + mb + v;
          dst[(size_t)hd * T_ + t] = (_Float16)acc[mt][f][v];
        }
      }
  }
}

// ---------------------------------------------------------------------------
// Kernel 2: banded QK^T -> sigmoid row stats.  Grid (H, T/128), 4 waves,
// 32 query rows per wave; K fragments shared by both M-tiles, ping-ponged.
// ---------------------------------------------------------------------------
#define PA_LOADK(KB0, KB1, S0)                                                 \
  {                                                                            \
    KB0 = load_b_cm(kbase + (size_t)(S0) * HD_, HD_);                          \
    KB1 = load_b_cm(kbase + (size_t)(S0) * HD_ + 32, HD_);                     \
  }
#define PA_COMPUTE(KB0, KB1, S0)                                               \
  {                                                                            \
    _Pragma("unroll") for (int mt = 0; mt < 2; ++mt) {                         \
      v8f s = v8f{};                                                           \
      s = wmma_f16(qa[mt][0], KB0, s);                                         \
      s = wmma_f16(qa[mt][1], KB1, s);                                         \
      _Pragma("unroll") for (int v = 0; v < 8; ++v) {                          \
        int row = t0 + mt * 16 + mb + v;                                       \
        int col = (S0) + nl;                                                   \
        if (col <= row && col >= row - WIN_) {                                 \
          float a = s[v] * 0.125f; /* 1/sqrt(HD) */                            \
          sig[mt][v] += 1.0f / (1.0f + expf(-a));                              \
        }                                                                      \
      }                                                                        \
    }                                                                          \
  }

__global__ __launch_bounds__(128) void mhga_phasea(
    const _Float16* __restrict__ Qh, const _Float16* __restrict__ Kh,
    float* __restrict__ denom, float* __restrict__ suminv) {
  int h = blockIdx.x;
  int wave = threadIdx.x >> 5, lane = threadIdx.x & 31;
  int t0 = blockIdx.y * 128 + wave * 32;
  int nl = lane & 15, mb = (lane >> 4) * 8;

  const _Float16* qp = Qh + (size_t)h * T_ * HD_ + (size_t)t0 * HD_;
  v16h qa[2][2];
#pragma unroll
  for (int mt = 0; mt < 2; ++mt) {
    qa[mt][0] = load_a_rm(qp + (size_t)mt * 16 * HD_, HD_);
    qa[mt][1] = load_a_rm(qp + (size_t)mt * 16 * HD_ + 32, HD_);
  }

  int s_lo = blockIdx.y * 128 - WIN_; if (s_lo < 0) s_lo = 0;
  int s_hi = blockIdx.y * 128 + 128;   // (s_hi - s_lo) is a multiple of 64

  const _Float16* kbase = Kh + (size_t)h * T_ * HD_;
  float sig[2][8];
#pragma unroll
  for (int mt = 0; mt < 2; ++mt)
#pragma unroll
    for (int v = 0; v < 8; ++v) sig[mt][v] = 0.0f;

  v16h kP0, kP1, kQ0, kQ1;
  PA_LOADK(kP0, kP1, s_lo);
  int s0 = s_lo;
  for (; s0 < s_hi - 32; s0 += 32) {
    PA_LOADK(kQ0, kQ1, s0 + 16);
    PA_COMPUTE(kP0, kP1, s0);
    PA_LOADK(kP0, kP1, s0 + 32);
    PA_COMPUTE(kQ0, kQ1, s0 + 16);
  }
  PA_LOADK(kQ0, kQ1, s0 + 16);
  PA_COMPUTE(kP0, kP1, s0);
  PA_COMPUTE(kQ0, kQ1, s0 + 16);

  float inv_acc = 0.0f;
#pragma unroll
  for (int mt = 0; mt < 2; ++mt)
#pragma unroll
    for (int v = 0; v < 8; ++v) {
      float r = hsum16(sig[mt][v]);
      int row = t0 + mt * 16 + mb + v;
      float bandc = (float)(((row < WIN_) ? row : WIN_) + 1);
      float gm = (r + 0.5f * ((float)T_ - bandc)) * (1.0f / (float)T_);
      float dn = gm + 0.5f;
      if (nl == 0) denom[h * T_ + row] = dn;
      inv_acc += 1.0f / dn;
    }
  if (nl == 0) atomicAdd(&suminv[h], inv_acc);
}

// ---------------------------------------------------------------------------
// Kernel 3: fitness-weighted softmax + PV.  Grid (H, T/128), 4 waves,
// 32 query rows per wave.  Logits are ~1e-3 -> exp without max-subtraction is
// exact softmax (out-of-band logits are -1e10 -> exp == 0 -> skipped).
// P goes C-layout -> wave-private LDS (f16) -> A-layout for the PV WMMA.
// ---------------------------------------------------------------------------
__global__ __launch_bounds__(128) void mhga_phaseb(
    const _Float16* __restrict__ Qh, const _Float16* __restrict__ Kh,
    const _Float16* __restrict__ Vt, const float* __restrict__ denom,
    const float* __restrict__ suminv, _Float16* __restrict__ Oh) {
  __shared__ __align__(32) _Float16 pb[4][2][16][40];  // per-wave 2x(16x32) P
  int h = blockIdx.x;
  int wave = threadIdx.x >> 5, lane = threadIdx.x & 31;
  int t0 = blockIdx.y * 128 + wave * 32;
  int nl = lane & 15, mb = (lane >> 4) * 8;

  const _Float16* qp = Qh + (size_t)h * T_ * HD_ + (size_t)t0 * HD_;
  v16h qa[2][2];
#pragma unroll
  for (int mt = 0; mt < 2; ++mt) {
    qa[mt][0] = load_a_rm(qp + (size_t)mt * 16 * HD_, HD_);
    qa[mt][1] = load_a_rm(qp + (size_t)mt * 16 * HD_ + 32, HD_);
  }

  float si = suminv[h];
  float fit[2][8];
#pragma unroll
  for (int mt = 0; mt < 2; ++mt)
#pragma unroll
    for (int v = 0; v < 8; ++v)
      fit[mt][v] = 0.125f / (denom[h * T_ + t0 + mt * 16 + mb + v] * si);

  int hv = h / (NH_ / HV_);
  const _Float16* vp = Vt + (size_t)hv * HD_ * T_;
  const _Float16* kbase = Kh + (size_t)h * T_ * HD_;

  v8f o[2][4];
#pragma unroll
  for (int mt = 0; mt < 2; ++mt)
#pragma unroll
    for (int dt = 0; dt < 4; ++dt) o[mt][dt] = v8f{};
  float rs[2][8];
#pragma unroll
  for (int mt = 0; mt < 2; ++mt)
#pragma unroll
    for (int v = 0; v < 8; ++v) rs[mt][v] = 0.0f;

  int s_lo = blockIdx.y * 128 - WIN_; if (s_lo < 0) s_lo = 0;
  int s_hi = blockIdx.y * 128 + 128;

  for (int c0 = s_lo; c0 < s_hi; c0 += 32) {
#pragma unroll
    for (int st = 0; st < 2; ++st) {
      int s0 = c0 + st * 16;
      v16h kb0 = load_b_cm(kbase + (size_t)s0 * HD_, HD_);
      v16h kb1 = load_b_cm(kbase + (size_t)s0 * HD_ + 32, HD_);
#pragma unroll
      for (int mt = 0; mt < 2; ++mt) {
        v8f s = v8f{};
        s = wmma_f16(qa[mt][0], kb0, s);
        s = wmma_f16(qa[mt][1], kb1, s);
#pragma unroll
        for (int v = 0; v < 8; ++v) {
          int row = t0 + mt * 16 + mb + v;
          int col = s0 + nl;
          float p = 0.0f;
          if (col <= row && col >= row - WIN_) {
            p = expf(s[v] * fit[mt][v]);
            rs[mt][v] += p;
          }
          pb[wave][mt][mb + v][st * 16 + nl] = (_Float16)p;
        }
      }
    }
    // same-wave LDS RAW: compiler inserts the s_wait_dscnt
#pragma unroll
    for (int dt = 0; dt < 4; ++dt) {
      v16h vb = load_b_cm(vp + (size_t)(dt * 16) * T_ + c0, T_);
#pragma unroll
      for (int mt = 0; mt < 2; ++mt) {
        v16h pa = load_a_rm(&pb[wave][mt][0][0], 40);
        o[mt][dt] = wmma_f16(pa, vb, o[mt][dt]);
      }
    }
  }

#pragma unroll
  for (int mt = 0; mt < 2; ++mt) {
    float rinv[8];
#pragma unroll
    for (int v = 0; v < 8; ++v) rinv[v] = 1.0f / hsum16(rs[mt][v]);
#pragma unroll
    for (int dt = 0; dt < 4; ++dt) {
      int d = dt * 16 + nl;
#pragma unroll
      for (int v = 0; v < 8; ++v) {
        int row = t0 + mt * 16 + mb + v;
        Oh[(size_t)row * D_ + h * HD_ + d] = (_Float16)(o[mt][dt][v] * rinv[v]);
      }
    }
  }
}

// ---------------------------------------------------------------------------
// Kernel 4: output projection  out = Oh(f16) @ WoT(f16) + b_o  -> fp32.
// 32x64 per wave, ping-pong pipelined like mhga_proj.
// ---------------------------------------------------------------------------
__global__ __launch_bounds__(128) void mhga_outproj(
    const _Float16* __restrict__ Oh, const _Float16* __restrict__ WoT,
    const float* __restrict__ bo, float* __restrict__ out) {
  int n0 = blockIdx.x * 64;
  int wave = threadIdx.x >> 5, lane = threadIdx.x & 31;
  int t0 = blockIdx.y * 128 + wave * 32;
  int nl = lane & 15, mb = (lane >> 4) * 8;

  const _Float16* xp0 = Oh + (size_t)t0 * D_;
  const _Float16* xp1 = Oh + (size_t)(t0 + 16) * D_;
  const _Float16* wp  = WoT + (size_t)n0 * D_;

  v8f acc[2][4];
#pragma unroll
  for (int mt = 0; mt < 2; ++mt)
#pragma unroll
    for (int f = 0; f < 4; ++f) acc[mt][f] = v8f{};

  v16h aP0, aP1, bP[4], aQ0, aQ1, bQ[4];
  GEMM_LOAD(aP0, aP1, bP, 0);
  int kk = 0;
  for (; kk < D_ - 64; kk += 64) {
    GEMM_LOAD(aQ0, aQ1, bQ, kk + 32);
    GEMM_STEP(aP0, aP1, bP);
    GEMM_LOAD(aP0, aP1, bP, kk + 64);
    GEMM_STEP(aQ0, aQ1, bQ);
  }
  GEMM_LOAD(aQ0, aQ1, bQ, kk + 32);
  GEMM_STEP(aP0, aP1, bP);
  GEMM_STEP(aQ0, aQ1, bQ);

#pragma unroll
  for (int f = 0; f < 4; ++f) {
    int n = n0 + f * 16 + nl;
    float bb = bo[n];
#pragma unroll
    for (int mt = 0; mt < 2; ++mt)
#pragma unroll
      for (int v = 0; v < 8; ++v) {
        int row = t0 + mt * 16 + mb + v;
        out[(size_t)row * D_ + n] = acc[mt][f][v] + bb;
      }
  }
}

// ---------------------------------------------------------------------------
// Workspace layout (all offsets 256B-aligned); total ~24.8 MB.
// ---------------------------------------------------------------------------
static const size_t OFF_WHT = 0;                                   // 2304*1024 f16
static const size_t OFF_WOT = OFF_WHT + (size_t)NQK_ * D_ * 2;     // 1024*1024 f16
static const size_t OFF_XH  = OFF_WOT + (size_t)D_ * D_ * 2;       // 2048*1024 f16
static const size_t OFF_QH  = OFF_XH + (size_t)T_ * D_ * 2;        // 16*2048*64 f16
static const size_t OFF_KH  = OFF_QH + (size_t)NH_ * T_ * HD_ * 2;
static const size_t OFF_VT  = OFF_KH + (size_t)NH_ * T_ * HD_ * 2; // 4*64*2048 f16
static const size_t OFF_OH  = OFF_VT + (size_t)HV_ * HD_ * T_ * 2; // 2048*1024 f16
static const size_t OFF_DEN = OFF_OH + (size_t)T_ * D_ * 2;        // 16*2048 f32
static const size_t OFF_SI  = OFF_DEN + (size_t)NH_ * T_ * 4;      // 16 f32

extern "C" void kernel_launch(void* const* d_in, const int* in_sizes, int n_in,
                              void* d_out, int out_size, void* d_ws, size_t ws_size,
                              hipStream_t stream) {
  (void)in_sizes; (void)n_in; (void)out_size; (void)ws_size;
  const float* x   = (const float*)d_in[0];
  const float* wq  = (const float*)d_in[1];
  const float* bq  = (const float*)d_in[2];
  const float* wk  = (const float*)d_in[3];
  const float* bk  = (const float*)d_in[4];
  const float* wv  = (const float*)d_in[5];
  const float* bv  = (const float*)d_in[6];
  const float* rqw = (const float*)d_in[7];
  const float* rkw = (const float*)d_in[8];
  const float* wo  = (const float*)d_in[9];
  const float* bo  = (const float*)d_in[10];
  float* out = (float*)d_out;

  char* ws = (char*)d_ws;
  _Float16* WhT  = (_Float16*)(ws + OFF_WHT);
  _Float16* WoT  = (_Float16*)(ws + OFF_WOT);
  _Float16* Xh   = (_Float16*)(ws + OFF_XH);
  _Float16* Qh   = (_Float16*)(ws + OFF_QH);
  _Float16* Kh   = (_Float16*)(ws + OFF_KH);
  _Float16* Vt   = (_Float16*)(ws + OFF_VT);
  _Float16* Oh   = (_Float16*)(ws + OFF_OH);
  float* denom   = (float*)(ws + OFF_DEN);
  float* suminv  = (float*)(ws + OFF_SI);

  mhga_wcvt<<<dim3(2048), 256, 0, stream>>>(x, wq, wk, wv, wo, WhT, WoT, Xh, suminv);
  mhga_proj<<<dim3(NQK_ / 64, T_ / 128), 128, 0, stream>>>(
      Xh, WhT, bq, bk, bv, rqw, rkw, Qh, Kh, Vt);
  mhga_phasea<<<dim3(NH_, T_ / 128), 128, 0, stream>>>(Qh, Kh, denom, suminv);
  mhga_phaseb<<<dim3(NH_, T_ / 128), 128, 0, stream>>>(Qh, Kh, Vt, denom, suminv, Oh);
  mhga_outproj<<<dim3(D_ / 64, T_ / 128), 128, 0, stream>>>(Oh, WoT, bo, out);
}
